// CFP_Encoder_12000138625424
// MI455X (gfx1250) — matrix operand order repaired and checked by reading, not compiled, not run on hardware
//
#include <hip/hip_runtime.h>

typedef __attribute__((ext_vector_type(16))) __bf16 v16bf;
typedef __attribute__((ext_vector_type(8)))  __bf16 v8bf;
typedef __attribute__((ext_vector_type(8)))  float  v8f;
typedef int v4i __attribute__((vector_size(16)));   // matches async builtin param type

__device__ __forceinline__ float siluf(float x) {
  return x * (1.f / (1.f + __expf(-x)));
}

// ---- CDNA5 async global->LDS copy (ASYNCcnt path), with safe fallback ------
#if __has_builtin(__builtin_amdgcn_global_load_async_to_lds_b128)
#define HAVE_ASYNC_LDS 1
#else
#define HAVE_ASYNC_LDS 0
#endif

#if __has_builtin(__builtin_amdgcn_s_wait_asynccnt)
#define WAIT_ASYNC(n) __builtin_amdgcn_s_wait_asynccnt(n)
#else
#define WAIT_ASYNC(n) asm volatile("s_wait_asynccnt %0" :: "i"(n) : "memory")
#endif

__device__ __forceinline__ void copy16_g2l(const __bf16* g, __bf16* l) {
#if HAVE_ASYNC_LDS
  __builtin_amdgcn_global_load_async_to_lds_b128((v4i*)g, (v4i*)l, 0, 0);
#else
  *(uint4*)l = *(const uint4*)g;
#endif
}

// ---------------------------------------------------------------------------
// Generic converters / elementwise
// ---------------------------------------------------------------------------
__global__ void cvt_bf16_kernel(const float* __restrict__ in, __bf16* __restrict__ out, int n) {
  for (int i = blockIdx.x * blockDim.x + threadIdx.x; i < n; i += gridDim.x * blockDim.x)
    out[i] = (__bf16)in[i];
}

__global__ void add_kernel(const float* __restrict__ a, const float* __restrict__ b,
                           float* __restrict__ o, int n) {
  int i = blockIdx.x * blockDim.x + threadIdx.x;
  if (i < n) o[i] = a[i] + b[i];
}

// ---------------------------------------------------------------------------
// WMMA GEMM: C[M,N](f32,ldc) = A[M,K](bf16,lda) @ Bw[K,N](bf16,ldb)
// optional +bias[N], +residual (f32, gathered by scatter row), row scatter.
// 256 threads = 8 waves; block tile 128x128, K-step 32; wave tile 32x64.
// Double-buffered LDS tiles streamed by global_load_async_to_lds_b128:
// ASYNCcnt completes in order, so wait<=pending keeps tile t resident while
// tile t+1 streams behind the 8 WMMAs. K must be a multiple of 32.
// ---------------------------------------------------------------------------
__global__ __launch_bounds__(256)
void wmma_gemm_kernel(const __bf16* __restrict__ A, int lda,
                      const __bf16* __restrict__ Bw, int ldb,
                      float* __restrict__ C, int ldc,
                      int M, int N, int K,
                      const float* __restrict__ bias,
                      const float* __restrict__ residual, int ldres,
                      const int* __restrict__ scatter, int Lseq)
{
  __shared__ __align__(16) __bf16 sA[2][128][40];  // pad: 20*m mod 64 distinct
  __shared__ __align__(16) __bf16 sB[2][32][136];  // pad: <=2-way on 16B reads
  const int tid  = threadIdx.x;
  const int lane = tid & 31;
  const int wave = tid >> 5;
  const int wm   = wave & 3;    // M slab: rows [wm*32, wm*32+32)
  const int wn   = wave >> 2;   // N slab: cols [wn*64, wn*64+64)
  const int blockM = blockIdx.y * 128;
  const int blockN = blockIdx.x * 128;
  const bool fullM = (blockM + 128) <= M;
  const bool fullN = (blockN + 128) <= N;
  const int half = lane >> 4;       // A fragment half
  const int mrow = lane & 15;

  v8f acc[2][4];
#pragma unroll
  for (int i = 0; i < 2; ++i)
#pragma unroll
    for (int j = 0; j < 4; ++j)
#pragma unroll
      for (int r = 0; r < 8; ++r) acc[i][j][r] = 0.f;

  // per-thread async chunks issued per tile (block-uniform)
  const int pend = (fullM ? 2 : 0) + (fullN ? 2 : 0);

  auto stage = [&](int k0, int buf) {
    // ---- A tile (128x32 bf16, row-major) ----
    if (fullM) {
#pragma unroll
      for (int j = 0; j < 2; ++j) {
        int idx = j * 256 + tid;            // 0..511 16B chunks
        int r = idx >> 2, c4 = idx & 3;
        copy16_g2l(A + (long)(blockM + r) * lda + k0 + c4 * 8, &sA[buf][r][c4 * 8]);
      }
    } else {
      for (int j = 0; j < 16; ++j) {
        int idx = j * 256 + tid;
        int r = idx >> 5, c = idx & 31;
        int gm = blockM + r;
        sA[buf][r][c] = (gm < M) ? A[(long)gm * lda + k0 + c] : (__bf16)0.f;
      }
    }
    // ---- B tile (32x128 bf16, row-major) ----
    if (fullN) {
#pragma unroll
      for (int j = 0; j < 2; ++j) {
        int idx = j * 256 + tid;            // 0..511 16B chunks
        int r = idx >> 4, c = idx & 15;
        copy16_g2l(Bw + (long)(k0 + r) * ldb + blockN + c * 8, &sB[buf][r][c * 8]);
      }
    } else {
      for (int j = 0; j < 16; ++j) {
        int idx = j * 256 + tid;
        int r = idx >> 7, n = idx & 127;
        int gn = blockN + n;
        sB[buf][r][n] = (gn < N) ? Bw[(long)(k0 + r) * ldb + gn] : (__bf16)0.f;
      }
    }
  };

  const int T = K >> 5;     // number of K tiles
  stage(0, 0);              // prologue: tile 0 -> buffer 0

  for (int t = 0; t < T; ++t) {
    const int cur = t & 1;
    const bool havenext = (t + 1) < T;
    if (havenext) stage((t + 1) << 5, cur ^ 1);  // stream next tile behind compute
    if (t + 2 < T) {  // run prefetch ahead of the async engine (global_prefetch_b8)
      __builtin_prefetch(A + (long)(blockM + (tid >> 1)) * lda + ((t + 2) << 5), 0, 0);
      __builtin_prefetch(Bw + (long)(((t + 2) << 5) + (tid >> 4)) * ldb + blockN + ((tid & 15) << 3), 0, 0);
    }
#if HAVE_ASYNC_LDS
    // wait for tile t's chunks; leave tile t+1's (pend per thread) in flight
    if (havenext) { if (pend == 4) WAIT_ASYNC(4); else if (pend == 2) WAIT_ASYNC(2); else WAIT_ASYNC(0); }
    else WAIT_ASYNC(0);
#endif
    __syncthreads();   // all waves' tile-t data resident

    // ---- fragments ----
    // A frag: lane l: row = l%16 of row-tile, half = l/16;
    //   elems 0..7 -> k = half*8+q ; elems 8..15 -> k = 16+half*8+q
    v16bf af[2], bfv[4];
#pragma unroll
    for (int i = 0; i < 2; ++i) {
      const __bf16* ar = &sA[cur][(wm * 2 + i) * 16 + mrow][half * 8];
      v8bf lo = *(const v8bf*)ar;
      v8bf hi = *(const v8bf*)(ar + 16);
#pragma unroll
      for (int q = 0; q < 8; ++q) { af[i][q] = lo[q]; af[i][8 + q] = hi[q]; }
    }
    // B frag: lane l: col = l%16 of col-tile, half = l/16;
    //   elems 0..7 -> k = half*8+q ; elems 8..15 -> k = 16+half*8+q
#pragma unroll
    for (int j = 0; j < 4; ++j) {
      const int bc = (wn * 4 + j) * 16 + mrow;
#pragma unroll
      for (int q = 0; q < 8; ++q) { bfv[j][q] = sB[cur][half * 8 + q][bc]; bfv[j][8 + q] = sB[cur][16 + half * 8 + q][bc]; }
    }
#pragma unroll
    for (int i = 0; i < 2; ++i)
#pragma unroll
      for (int j = 0; j < 4; ++j)
        acc[i][j] = __builtin_amdgcn_wmma_f32_16x16x32_bf16(
            false, af[i], false, bfv[j], (short)0, acc[i][j], false, false);

    __syncthreads();   // all waves done reading buf[cur] before it is restaged
  }

  // ---- epilogue: C/D layout: vgpr r: lanes0-15 M=r, lanes16-31 M=r+8; N=lane%16
#pragma unroll
  for (int i = 0; i < 2; ++i) {
    int mt = blockM + wm * 32 + i * 16;
#pragma unroll
    for (int j = 0; j < 4; ++j) {
      int nt = blockN + wn * 64 + j * 16;
      int gn = nt + (lane & 15);
      int mbase = mt + ((lane >> 4) << 3);
#pragma unroll
      for (int r = 0; r < 8; ++r) {
        int gm = mbase + r;
        if (gm < M && gn < N) {
          float v = acc[i][j][r];
          if (bias) v += bias[gn];
          int orow = gm;
          if (scatter) { int bb = gm / Lseq; int l = gm - bb * Lseq; orow = bb * Lseq + scatter[l]; }
          if (residual) v += residual[(long)orow * ldres + gn];
          C[(long)orow * ldc + gn] = v;
        }
      }
    }
  }
}

// ---------------------------------------------------------------------------
// LayerNorm with optional gather (permutation); writes bf16 for the GEMM.
// One block (256 thr) per row; D in {512,1024}.
// ---------------------------------------------------------------------------
__global__ __launch_bounds__(256)
void ln_kernel(const float* __restrict__ in, int ldin,
               __bf16* __restrict__ out, int ldout,
               const float* __restrict__ w, const float* __restrict__ b,
               int D, const int* __restrict__ gather, int Lseq)
{
  __shared__ float red[256];
  int m = blockIdx.x;
  int src = m;
  if (gather) { int bb = m / Lseq; int l = m - bb * Lseq; src = bb * Lseq + gather[l]; }
  const float* row = in + (long)src * ldin;
  int nv = D >> 8;  // D/256 (<=4)
  float v[4];
  float s = 0.f;
  for (int j = 0; j < nv; ++j) { v[j] = row[(j << 8) + threadIdx.x]; s += v[j]; }
  red[threadIdx.x] = s; __syncthreads();
  for (int off = 128; off > 0; off >>= 1) {
    if (threadIdx.x < off) red[threadIdx.x] += red[threadIdx.x + off];
    __syncthreads();
  }
  float mean = red[0] / (float)D;
  __syncthreads();
  float ss = 0.f;
  for (int j = 0; j < nv; ++j) { float d = v[j] - mean; ss += d * d; }
  red[threadIdx.x] = ss; __syncthreads();
  for (int off = 128; off > 0; off >>= 1) {
    if (threadIdx.x < off) red[threadIdx.x] += red[threadIdx.x + off];
    __syncthreads();
  }
  float rstd = rsqrtf(red[0] / (float)D + 1e-5f);
  __bf16* orow = out + (long)m * ldout;
  for (int j = 0; j < nv; ++j) {
    int d = (j << 8) + threadIdx.x;
    orow[d] = (__bf16)((v[j] - mean) * rstd * w[d] + b[d]);
  }
}

// ---------------------------------------------------------------------------
// Front end
// ---------------------------------------------------------------------------
__global__ void front_max_kernel(const float* __restrict__ x,
                                 const float* __restrict__ c3w, const float* __restrict__ c3b,
                                 const float* __restrict__ c5w, const float* __restrict__ c5b,
                                 const float* __restrict__ c7w, const float* __restrict__ c7b,
                                 const float* __restrict__ pcw, const float* __restrict__ pcb,
                                 float* __restrict__ out)
{
  int t = blockIdx.x * blockDim.x + threadIdx.x;
  if (t >= 4 * 64 * 64) return;
  int xx = t & 63, yy = (t >> 6) & 63, b = t >> 12;
  float cat[12];
  for (int co = 0; co < 4; ++co) {
    float a = c3b[co];
    for (int ci = 0; ci < 4; ++ci)
      for (int ky = 0; ky < 3; ++ky) {
        int iy = yy - 1 + ky; if (iy < 0 || iy > 63) continue;
        for (int kx = 0; kx < 3; ++kx) {
          int ix = xx - 1 + kx; if (ix < 0 || ix > 63) continue;
          a += c3w[((co * 4 + ci) * 3 + ky) * 3 + kx] * x[((b * 4 + ci) * 64 + iy) * 64 + ix];
        }
      }
    cat[co] = a;
  }
  for (int co = 0; co < 4; ++co) {
    float a = c5b[co];
    for (int ci = 0; ci < 4; ++ci)
      for (int ky = 0; ky < 5; ++ky) {
        int iy = yy - 2 + ky; if (iy < 0 || iy > 63) continue;
        for (int kx = 0; kx < 5; ++kx) {
          int ix = xx - 2 + kx; if (ix < 0 || ix > 63) continue;
          a += c5w[((co * 4 + ci) * 5 + ky) * 5 + kx] * x[((b * 4 + ci) * 64 + iy) * 64 + ix];
        }
      }
    cat[4 + co] = a;
  }
  for (int co = 0; co < 4; ++co) {
    float a = c7b[co];
    for (int ci = 0; ci < 4; ++ci)
      for (int ky = 0; ky < 7; ++ky) {
        int iy = yy - 3 + ky; if (iy < 0 || iy > 63) continue;
        for (int kx = 0; kx < 7; ++kx) {
          int ix = xx - 3 + kx; if (ix < 0 || ix > 63) continue;
          a += c7w[((co * 4 + ci) * 7 + ky) * 7 + kx] * x[((b * 4 + ci) * 64 + iy) * 64 + ix];
        }
      }
    cat[8 + co] = a;
  }
  for (int o = 0; o < 4; ++o) {
    float a = pcb[o];
    for (int ci = 0; ci < 12; ++ci) a += pcw[o * 12 + ci] * cat[ci];
    out[((b * 4 + o) * 64 + yy) * 64 + xx] = a;
  }
}

// adaptive avg pools to 3x3 / 5x5 / 7x7; packed: [p3(16*9)][p5(16*25)][p7(16*49)]
__global__ void pool_kernel(const float* __restrict__ x, float* __restrict__ pooled)
{
  int t = blockIdx.x * blockDim.x + threadIdx.x;
  if (t >= 1328) return;
  int s, base, idx;
  if (t < 144)      { s = 3; base = 0;   idx = t; }
  else if (t < 544) { s = 5; base = 144; idx = t - 144; }
  else              { s = 7; base = 544; idx = t - 544; }
  int ss = s * s;
  int bc = idx / ss, ij = idx - bc * ss;
  int i = ij / s, j = ij - i * s;
  int b = bc >> 2, c = bc & 3;
  int y0 = i * 64 / s, y1 = ((i + 1) * 64 + s - 1) / s;
  int x0 = j * 64 / s, x1 = ((j + 1) * 64 + s - 1) / s;
  float sum = 0.f;
  for (int y = y0; y < y1; ++y)
    for (int xp = x0; xp < x1; ++xp)
      sum += x[((b * 4 + c) * 64 + y) * 64 + xp];
  pooled[base + bc * ss + ij] = sum / (float)((y1 - y0) * (x1 - x0));
}

__global__ void front_avg_kernel(const float* __restrict__ pooled,
                                 const float* __restrict__ apw, const float* __restrict__ apb,
                                 float* __restrict__ out)
{
  int t = blockIdx.x * blockDim.x + threadIdx.x;
  if (t >= 4 * 64 * 64) return;
  int xx = t & 63, yy = (t >> 6) & 63, b = t >> 12;
  const int sizes[3] = {3, 5, 7};
  const int bases[3] = {0, 144, 544};
  float cat[12];
  for (int si = 0; si < 3; ++si) {
    int s = sizes[si], base = bases[si];
    float srcy = fmaxf((yy + 0.5f) * s / 64.f - 0.5f, 0.f);
    int ly = (int)srcy; int hy = ly + 1 < s ? ly + 1 : s - 1; float wy = srcy - ly;
    float srcx = fmaxf((xx + 0.5f) * s / 64.f - 0.5f, 0.f);
    int lx = (int)srcx; int hx = lx + 1 < s ? lx + 1 : s - 1; float wx = srcx - lx;
    for (int c = 0; c < 4; ++c) {
      const float* p = pooled + base + (b * 4 + c) * s * s;
      float v0 = p[ly * s + lx] * (1.f - wy) + p[hy * s + lx] * wy;
      float v1 = p[ly * s + hx] * (1.f - wy) + p[hy * s + hx] * wy;
      cat[si * 4 + c] = v0 * (1.f - wx) + v1 * wx;
    }
  }
  for (int o = 0; o < 4; ++o) {
    float a = apb[o];
    for (int ci = 0; ci < 12; ++ci) a += apw[o * 12 + ci] * cat[ci];
    out[((b * 4 + o) * 64 + yy) * 64 + xx] = a;
  }
}

// patch embed: stride-2 2x2 conv over 4 chans + pos_embed; tok (4096 x 512) f32
__global__ void embed_kernel(const float* __restrict__ t,
                             const float* __restrict__ pew, const float* __restrict__ peb,
                             const float* __restrict__ pos, float* __restrict__ tok)
{
  int id = blockIdx.x * blockDim.x + threadIdx.x;
  if (id >= 4 * 1024 * 512) return;
  int d = id & 511, l = (id >> 9) & 1023, b = id >> 19;
  int gy = l >> 5, gx = l & 31;
  float a = peb[d];
  for (int c = 0; c < 4; ++c)
    for (int py = 0; py < 2; ++py)
      for (int px = 0; px < 2; ++px)
        a += pew[((d * 4 + c) * 2 + py) * 2 + px] *
             t[((b * 4 + c) * 64 + gy * 2 + py) * 64 + gx * 2 + px];
  tok[(long)(b * 1024 + l) * 512 + d] = a + pos[l * 512 + d];
}

// ---------------------------------------------------------------------------
// Mamba-block elementwise kernels
// ---------------------------------------------------------------------------
// causal depthwise conv (4 taps) along L + SiLU; writes f32 (scan) + bf16 (GEMM)
__global__ void dwconv_silu_kernel(const float* __restrict__ xz,
                                   const float* __restrict__ cw, const float* __restrict__ cb,
                                   float* __restrict__ xsA, __bf16* __restrict__ xsbf)
{
  int id = blockIdx.x * blockDim.x + threadIdx.x;
  if (id >= 4096 * 1024) return;
  int e = id & 1023, m = id >> 10, l = m & 1023;
  float a = cb[e];
#pragma unroll
  for (int k = 0; k < 4; ++k) {
    int li = l - 3 + k;
    if (li >= 0) a += cw[e * 4 + k] * xz[(long)(m - 3 + k) * 2048 + e];
  }
  float o = siluf(a);
  xsA[(long)m * 1024 + e]  = o;
  xsbf[(long)m * 1024 + e] = (__bf16)o;
}

// dt = softplus(dbl[:, :16] @ W_dt + b_dt)   (K=16: VALU is cheapest)
__global__ void dt_softplus_kernel(const float* __restrict__ dbl,
                                   const float* __restrict__ wdt, const float* __restrict__ bdt,
                                   float* __restrict__ dt)
{
  int id = blockIdx.x * blockDim.x + threadIdx.x;
  if (id >= 4096 * 1024) return;
  int e = id & 1023, m = id >> 10;
  const float* dr = dbl + (long)m * 48;
  float a = bdt[e];
#pragma unroll
  for (int r = 0; r < 16; ++r) a += dr[r] * wdt[r * 1024 + e];
  dt[(long)m * 1024 + e] = a > 20.f ? a : log1pf(__expf(a));
}

// selective scan: one thread per (b, e) channel; 16 states unrolled.
__global__ __launch_bounds__(256)
void scan_kernel(const float* __restrict__ dt, const float* __restrict__ xs,
                 const float* __restrict__ dbl, const float* __restrict__ Alog,
                 float* __restrict__ ys)
{
  int e = ((blockIdx.x & 3) << 8) + threadIdx.x;  // 0..1023
  int b = blockIdx.x >> 2;                        // 0..3
  float Ae[16], h[16];
#pragma unroll
  for (int n = 0; n < 16; ++n) { Ae[n] = -__expf(Alog[e * 16 + n]); h[n] = 0.f; }
  const int base = b * 1024;
  for (int l = 0; l < 1024; ++l) {
    int m = base + l;
    float dv = dt[(long)m * 1024 + e];
    float dx = dv * xs[(long)m * 1024 + e];
    const float* dr = dbl + (long)m * 48;
    float acc = 0.f;
#pragma unroll
    for (int n = 0; n < 16; ++n) {
      h[n] = __expf(dv * Ae[n]) * h[n] + dx * dr[16 + n];
      acc += h[n] * dr[32 + n];
    }
    ys[(long)m * 1024 + e] = acc;
  }
}

// y = (ys + xs*Dp) * silu(res);  res at xz col 1024+e ; bf16 out for GEMM3
__global__ void gate_kernel(const float* __restrict__ ys, const float* __restrict__ xs,
                            const float* __restrict__ dp, const float* __restrict__ xz,
                            __bf16* __restrict__ y)
{
  int id = blockIdx.x * blockDim.x + threadIdx.x;
  if (id >= 4096 * 1024) return;
  int e = id & 1023, m = id >> 10;
  float r = xz[(long)m * 2048 + 1024 + e];
  float o = (ys[(long)m * 1024 + e] + xs[(long)m * 1024 + e] * dp[e]) * siluf(r);
  y[(long)m * 1024 + e] = (__bf16)o;
}

// ---------------------------------------------------------------------------
// Launcher
// ---------------------------------------------------------------------------
extern "C" void kernel_launch(void* const* d_in, const int* in_sizes, int n_in,
                              void* d_out, int out_size, void* d_ws, size_t ws_size,
                              hipStream_t stream)
{
  (void)in_sizes; (void)n_in; (void)out_size; (void)ws_size;
  const int Mtok = 4096, Dm = 512, Em = 1024, Ls = 1024, NB = 16;

  // ---- input indices (setup_inputs dict order, nested dicts flattened) ----
  const float* x    = (const float*)d_in[0];
  const float* c3w  = (const float*)d_in[1];  const float* c3b = (const float*)d_in[2];
  const float* c5w  = (const float*)d_in[3];  const float* c5b = (const float*)d_in[4];
  const float* c7w  = (const float*)d_in[5];  const float* c7b = (const float*)d_in[6];
  const float* pc1w = (const float*)d_in[7];  const float* pc1b = (const float*)d_in[8];
  const float* ap1w = (const float*)d_in[9];  const float* ap1b = (const float*)d_in[10];
  const float* pew  = (const float*)d_in[11]; const float* peb  = (const float*)d_in[12];
  // zp: 13..23  vp: 24..34  (ln_w, ln_b, W_in, conv_w, conv_b, W_x, W_dt, b_dt, A_log, Dp, W_out)
  const float* pos  = (const float*)d_in[35];
  const float* hlnw = (const float*)d_in[36]; const float* hlnb = (const float*)d_in[37];
  const float* hW   = (const float*)d_in[38]; const float* hB   = (const float*)d_in[39];
  const int*   perm = (const int*)d_in[40];

  // ---- workspace carve ----
  char* wsp = (char*)d_ws;
  size_t off = 0;
  auto alloc = [&](size_t bytes) -> void* {
    void* p = wsp + off; off += (bytes + 255) & ~(size_t)255; return p;
  };
  __bf16* winbf[2];  winbf[0]  = (__bf16*)alloc((size_t)16 * 512 * 2048 * 2);
                     winbf[1]  = (__bf16*)alloc((size_t)16 * 512 * 2048 * 2);
  __bf16* wxbf[2];   wxbf[0]   = (__bf16*)alloc((size_t)16 * 1024 * 48 * 2);
                     wxbf[1]   = (__bf16*)alloc((size_t)16 * 1024 * 48 * 2);
  __bf16* woutbf[2]; woutbf[0] = (__bf16*)alloc((size_t)16 * 1024 * 512 * 2);
                     woutbf[1] = (__bf16*)alloc((size_t)16 * 1024 * 512 * 2);
  __bf16* wheadbf   = (__bf16*)alloc((size_t)1024 * 512 * 2);
  float* maxout = (float*)alloc((size_t)4 * 4 * 64 * 64 * 4);
  float* avgout = (float*)alloc((size_t)4 * 4 * 64 * 64 * 4);
  float* pooled = (float*)alloc((size_t)1328 * 4);
  float* tok    = (float*)alloc((size_t)Mtok * Dm * 4);
  float* outs   = (float*)alloc((size_t)16 * Mtok * Dm * 4);
  float* comb   = (float*)alloc((size_t)Mtok * 1024 * 4);
  float* inpbuf = (float*)alloc((size_t)Mtok * Dm * 4);
  __bf16* xnbf  = (__bf16*)alloc((size_t)Mtok * Dm * 2);
  float* xz     = (float*)alloc((size_t)Mtok * 2048 * 4);
  float* xsA    = (float*)alloc((size_t)Mtok * Em * 4);
  __bf16* xsbf  = (__bf16*)alloc((size_t)Mtok * Em * 2);
  float* dblb   = (float*)alloc((size_t)Mtok * 48 * 4);
  float* dtb    = (float*)alloc((size_t)Mtok * Em * 4);
  float* ysb    = (float*)alloc((size_t)Mtok * Em * 4);
  __bf16* ybf   = (__bf16*)alloc((size_t)Mtok * Em * 2);
  __bf16* hlnbf = (__bf16*)alloc((size_t)Mtok * 1024 * 2);

  auto cvt = [&](const float* src, __bf16* dst, long n) {
    long g = (n + 255) / 256; if (g > 8192) g = 8192;
    cvt_bf16_kernel<<<(int)g, 256, 0, stream>>>(src, dst, (int)n);
  };
  // bf16 weight conversion (deterministic, each launch)
  cvt((const float*)d_in[13 + 2],  winbf[0],  (long)16 * 512 * 2048);
  cvt((const float*)d_in[24 + 2],  winbf[1],  (long)16 * 512 * 2048);
  cvt((const float*)d_in[13 + 5],  wxbf[0],   (long)16 * 1024 * 48);
  cvt((const float*)d_in[24 + 5],  wxbf[1],   (long)16 * 1024 * 48);
  cvt((const float*)d_in[13 + 10], woutbf[0], (long)16 * 1024 * 512);
  cvt((const float*)d_in[24 + 10], woutbf[1], (long)16 * 1024 * 512);
  cvt(hW, wheadbf, (long)1024 * 512);

  // ---- front end ----
  front_max_kernel<<<64, 256, 0, stream>>>(x, c3w, c3b, c5w, c5b, c7w, c7b, pc1w, pc1b, maxout);
  pool_kernel<<<6, 256, 0, stream>>>(x, pooled);
  front_avg_kernel<<<64, 256, 0, stream>>>(pooled, ap1w, ap1b, avgout);

  // ---- two mamba stacks ----
  for (int s = 0; s < 2; ++s) {
    int pb = (s == 0) ? 13 : 24;
    const float* lnw   = (const float*)d_in[pb + 0];
    const float* lnb   = (const float*)d_in[pb + 1];
    const float* convw = (const float*)d_in[pb + 3];
    const float* convb = (const float*)d_in[pb + 4];
    const float* wdt   = (const float*)d_in[pb + 6];
    const float* bdt   = (const float*)d_in[pb + 7];
    const float* alog  = (const float*)d_in[pb + 8];
    const float* dpp   = (const float*)d_in[pb + 9];
    const __bf16* Winb  = winbf[s];
    const __bf16* Wxb   = wxbf[s];
    const __bf16* Woutb = woutbf[s];
    const int* permAll = (s == 0) ? perm : nullptr;

    embed_kernel<<<8192, 256, 0, stream>>>(s == 0 ? maxout : avgout, pew, peb, pos, tok);

    for (int i = 0; i < NB; ++i) {
      const float* inp;
      if (i == 0) inp = tok;
      else if (i <= 8) inp = outs + (size_t)(i - 1) * Mtok * Dm;
      else {
        add_kernel<<<8192, 256, 0, stream>>>(outs + (size_t)(i - 1) * Mtok * Dm,
                                             outs + (size_t)(15 - i) * Mtok * Dm,
                                             inpbuf, Mtok * Dm);
        inp = inpbuf;
      }
      const int* permRow = permAll ? permAll + i * Ls : nullptr;

      ln_kernel<<<Mtok, 256, 0, stream>>>(inp, Dm, xnbf, Dm, lnw + i * Dm, lnb + i * Dm,
                                          Dm, permRow, Ls);
      // xn @ W_in -> xz (4096 x 2048)
      wmma_gemm_kernel<<<dim3(16, 32), 256, 0, stream>>>(
          xnbf, Dm, Winb + (size_t)i * Dm * 2048, 2048, xz, 2048,
          Mtok, 2048, Dm, nullptr, nullptr, 0, nullptr, Ls);
      dwconv_silu_kernel<<<16384, 256, 0, stream>>>(xz, convw + (size_t)i * Em * 4,
                                                    convb + (size_t)i * Em, xsA, xsbf);
      // xs @ W_x -> dbl (4096 x 48)
      wmma_gemm_kernel<<<dim3(1, 32), 256, 0, stream>>>(
          xsbf, Em, Wxb + (size_t)i * Em * 48, 48, dblb, 48,
          Mtok, 48, Em, nullptr, nullptr, 0, nullptr, Ls);
      dt_softplus_kernel<<<16384, 256, 0, stream>>>(dblb, wdt + (size_t)i * 16 * Em,
                                                    bdt + (size_t)i * Em, dtb);
      scan_kernel<<<16, 256, 0, stream>>>(dtb, xsA, dblb, alog + (size_t)i * Em * 16, ysb);
      gate_kernel<<<16384, 256, 0, stream>>>(ysb, xsA, dpp + (size_t)i * Em, xz, ybf);

      float* outp; int ldo;
      if (i == NB - 1) { outp = comb + (s == 0 ? 0 : Dm); ldo = 2 * Dm; }
      else             { outp = outs + (size_t)i * Mtok * Dm; ldo = Dm; }
      // y @ W_out + inp, scattered through perm into outs/comb
      wmma_gemm_kernel<<<dim3(4, 32), 256, 0, stream>>>(
          ybf, Em, Woutb + (size_t)i * Em * Dm, Dm, outp, ldo,
          Mtok, Dm, Em, nullptr, inp, Dm, permRow, Ls);
    }
  }

  // ---- head: LN(1024) then (4096x1024)@(1024x512)+b -> d_out ----
  ln_kernel<<<Mtok, 256, 0, stream>>>(comb, 1024, hlnbf, 1024, hlnw, hlnb, 1024, nullptr, Ls);
  wmma_gemm_kernel<<<dim3(4, 32), 256, 0, stream>>>(
      hlnbf, 1024, wheadbf, 512, (float*)d_out, 512,
      Mtok, 512, 1024, hB, nullptr, 0, nullptr, Ls);
}
